// GazeLSTM_68762426409180
// MI455X (gfx1250) — matrix operand order repaired and checked by reading, not compiled, not run on hardware
//
#include <hip/hip_runtime.h>

typedef _Float16 v8h  __attribute__((ext_vector_type(8)));
typedef _Float16 v16h __attribute__((ext_vector_type(16)));
typedef float    v8f  __attribute__((ext_vector_type(8)));

#define HID   256
#define GATES 1024          // 4*HID
#define ROWS  32            // batch rows per workgroup (2 M-tiles of 16)
#define LSEQ  32
#define NOUT  128

__device__ __forceinline__ float fast_sigmoid(float x) {
    return 1.0f / (1.0f + __expf(-x));
}
__device__ __forceinline__ float fast_tanh(float x) {
    float e = __expf(-2.0f * x);
    return (1.0f - e) / (1.0f + e);
}

// A fragment (16x32 f16) from an LDS row-major [32][256] f16 tile.
// ISA 16-bit A layout: lanes 0-15 hold M=lane, K = {k0..k0+7, k0+16..k0+23};
// lanes 16-31 hold M=lane-16, K = {k0+8..k0+15, k0+24..k0+31}.
__device__ __forceinline__ v16h load_a_frag(const _Float16* hb, int mt, int k,
                                            int lane16, int hi) {
    const _Float16* base = hb + (mt * 16 + lane16) * HID + k * 32 + hi * 8;
    v8h lo = *(const v8h*)(base);
    v8h hx = *(const v8h*)(base + 16);
    v16h a;
#pragma unroll
    for (int i = 0; i < 8; ++i) { a[i] = lo[i]; a[i + 8] = hx[i]; }
    return a;
}

// B fragment (32x16 f16) for column `col` of a row-major [cols][256] f16 weight
// matrix (B element (k,n) = W[n][k], i.e. the GEMM is h @ W^T).
// B layout (matches A): lanes 0-15 hold N=lane, K={k0..k0+7, k0+16..k0+23}; lanes 16-31 K={k0+8..k0+15, k0+24..k0+31}.
__device__ __forceinline__ v16h load_b_frag(const _Float16* __restrict__ w,
                                            int col, int k, int hi) {
    const _Float16* base = w + (size_t)col * HID + k * 32 + hi * 8;
    v8h lo = *(const v8h*)(base);
    v8h hx = *(const v8h*)(base + 16);
    v16h b;
#pragma unroll
    for (int i = 0; i < 8; ++i) { b[i] = lo[i]; b[i + 8] = hx[i]; }
    return b;
}

__global__ void cvt_weights_f16(const float* __restrict__ a, _Float16* __restrict__ da, int na,
                                const float* __restrict__ b, _Float16* __restrict__ db, int nb) {
    int i = blockIdx.x * blockDim.x + threadIdx.x;
    if (i < na) da[i] = (_Float16)a[i];
    if (i < nb) db[i] = (_Float16)b[i];
}

__global__ __launch_bounds__(256)
void lstm_fused(const float* __restrict__ x,        // [N, L, 2]
                const float* __restrict__ W_ih,     // [1024, 2]
                const float* __restrict__ b_ih,     // [1024]
                const float* __restrict__ b_hh,     // [1024]
                const float* __restrict__ fc_b,     // [128]
                const _Float16* __restrict__ whh16, // [1024][256]
                const _Float16* __restrict__ fcw16, // [128][256]
                float* __restrict__ out)            // [N, L, 128]
{
    __shared__ _Float16 lds_h[2][ROWS * HID];   // double-buffered h, 32 KB
    __shared__ float    lds_x[ROWS * 2];        // staged x for current step

    const int tid    = threadIdx.x;
    const int wave   = tid >> 5;
    const int lane   = tid & 31;
    const int lane16 = lane & 15;
    const int hi     = (lane >= 16) ? 1 : 0;
    const size_t row0 = (size_t)blockIdx.x * ROWS;

    // Per-lane constants: wave owns h-columns [wave*32, wave*32+32) (2 N-tiles)
    float wih0[4][2], wih1[4][2], bsum[4][2];
#pragma unroll
    for (int g = 0; g < 4; ++g)
#pragma unroll
        for (int nt = 0; nt < 2; ++nt) {
            int gc = g * 256 + wave * 32 + nt * 16 + lane16;
            wih0[g][nt] = W_ih[gc * 2 + 0];
            wih1[g][nt] = W_ih[gc * 2 + 1];
            bsum[g][nt] = b_ih[gc] + b_hh[gc];
        }
    const float fcb = fc_b[wave * 16 + lane16];

    // h(0) = 0; stage x for step 0
    for (int i = tid; i < ROWS * HID; i += 256) lds_h[0][i] = (_Float16)0.0f;
    if (tid < ROWS * 2) {
        int r = tid >> 1, c = tid & 1;
        lds_x[tid] = x[((row0 + r) * LSEQ + 0) * 2 + c];
    }
    __syncthreads();

    const v8f zero8 = {0.f, 0.f, 0.f, 0.f, 0.f, 0.f, 0.f, 0.f};
    v8f cst[2][2];  // cell state: [M-tile][N-tile], C/D register layout
#pragma unroll
    for (int mt = 0; mt < 2; ++mt)
#pragma unroll
        for (int nt = 0; nt < 2; ++nt) cst[mt][nt] = zero8;

#pragma unroll 1
    for (int l = 0; l < LSEQ; ++l) {
        const int p = l & 1;
        const _Float16* hb = &lds_h[p][0];

        // gates[32 rows x (4 gates x 32 cols)] = h @ W_hh^T  (f32 accum)
        v8f acc[2][8];
#pragma unroll
        for (int mt = 0; mt < 2; ++mt)
#pragma unroll
            for (int t = 0; t < 8; ++t) acc[mt][t] = zero8;

        // Runtime K loop: keeps W_hh B-fragments un-hoistable (no spills) while
        // batching all 8 B loads + 2 A loads ahead of the 16 WMMAs per iteration.
#pragma unroll 1
        for (int k = 0; k < 8; ++k) {
            v16h a0 = load_a_frag(hb, 0, k, lane16, hi);
            v16h a1 = load_a_frag(hb, 1, k, lane16, hi);
            v16h bfr[8];
#pragma unroll
            for (int t = 0; t < 8; ++t) {
                int g  = t >> 1;
                int nt = t & 1;
                bfr[t] = load_b_frag(whh16, g * 256 + wave * 32 + nt * 16 + lane16, k, hi);
            }
#pragma unroll
            for (int t = 0; t < 8; ++t) {
                acc[0][t] = __builtin_amdgcn_wmma_f32_16x16x32_f16(
                    false, a0, false, bfr[t], (short)0, acc[0][t], false, false);
                acc[1][t] = __builtin_amdgcn_wmma_f32_16x16x32_f16(
                    false, a1, false, bfr[t], (short)0, acc[1][t], false, false);
            }
        }

        // nonlinearity, c/h update, write h(l+1) to the other LDS buffer
#pragma unroll
        for (int mt = 0; mt < 2; ++mt)
#pragma unroll
            for (int nt = 0; nt < 2; ++nt) {
                const int col = wave * 32 + nt * 16 + lane16;
#pragma unroll
                for (int v = 0; v < 8; ++v) {
                    int rl = mt * 16 + v + hi * 8;
                    float xs0 = lds_x[rl * 2 + 0];
                    float xs1 = lds_x[rl * 2 + 1];
                    float vi = acc[mt][0 + nt][v] + wih0[0][nt] * xs0 + wih1[0][nt] * xs1 + bsum[0][nt];
                    float vf = acc[mt][2 + nt][v] + wih0[1][nt] * xs0 + wih1[1][nt] * xs1 + bsum[1][nt];
                    float vg = acc[mt][4 + nt][v] + wih0[2][nt] * xs0 + wih1[2][nt] * xs1 + bsum[2][nt];
                    float vo = acc[mt][6 + nt][v] + wih0[3][nt] * xs0 + wih1[3][nt] * xs1 + bsum[3][nt];
                    float ig = fast_sigmoid(vi);
                    float fg = fast_sigmoid(vf);
                    float gg = fast_tanh(vg);
                    float og = fast_sigmoid(vo);
                    float cn = fg * cst[mt][nt][v] + ig * gg;
                    cst[mt][nt][v] = cn;
                    float hn = og * fast_tanh(cn);
                    lds_h[1 - p][rl * HID + col] = (_Float16)hn;
                }
            }

        __syncthreads();  // h(l+1) visible to everyone; lds_x reads of step l done

        // stage x for the next step (read at step l+1, after the barrier below)
        if (l + 1 < LSEQ && tid < ROWS * 2) {
            int r = tid >> 1, c = tid & 1;
            lds_x[tid] = x[((row0 + r) * LSEQ + (l + 1)) * 2 + c];
        }

        // fused FC: out(:, l, 16w..16w+16) = h(l+1) @ fc_w^T + fc_b
        // Runtime K loop again so the 8 fc_w fragments are never hoisted across l.
        const _Float16* hb2 = &lds_h[1 - p][0];
#pragma unroll
        for (int mt = 0; mt < 2; ++mt) {
            v8f oacc = zero8;
#pragma unroll 1
            for (int k = 0; k < 8; ++k) {
                v16h a = load_a_frag(hb2, mt, k, lane16, hi);
                v16h b = load_b_frag(fcw16, wave * 16 + lane16, k, hi);
                oacc = __builtin_amdgcn_wmma_f32_16x16x32_f16(
                    false, a, false, b, (short)0, oacc, false, false);
            }
#pragma unroll
            for (int v = 0; v < 8; ++v) {
                int rl = mt * 16 + v + hi * 8;
                size_t n = row0 + rl;
                out[(n * LSEQ + l) * NOUT + wave * 16 + lane16] = oacc[v] + fcb;
            }
        }

        __syncthreads();  // protect lds_x overwrite and lds_h rotation
    }
}

extern "C" void kernel_launch(void* const* d_in, const int* in_sizes, int n_in,
                              void* d_out, int out_size, void* d_ws, size_t ws_size,
                              hipStream_t stream) {
    const float* x    = (const float*)d_in[0];
    const float* W_ih = (const float*)d_in[1];
    const float* W_hh = (const float*)d_in[2];
    const float* b_ih = (const float*)d_in[3];
    const float* b_hh = (const float*)d_in[4];
    const float* fc_w = (const float*)d_in[5];
    const float* fc_b = (const float*)d_in[6];
    float* out = (float*)d_out;

    _Float16* whh16 = (_Float16*)d_ws;           // 1024*256 f16 = 512 KB
    _Float16* fcw16 = whh16 + GATES * HID;       // 128*256  f16 =  64 KB

    const int na = GATES * HID;   // 262144
    const int nb = NOUT * HID;    // 32768
    cvt_weights_f16<<<(na + 255) / 256, 256, 0, stream>>>(W_hh, whh16, na, fc_w, fcw16, nb);

    const int nrows = 64 * 128;   // B*T = 8192
    lstm_fused<<<nrows / ROWS, 256, 0, stream>>>(x, W_ih, b_ih, b_hh, fc_b,
                                                 whh16, fcw16, out);
}
